// DiagWinAttention_73701638800201
// MI455X (gfx1250) — matrix operand order repaired and checked by reading, not compiled, not run on hardware
//
#include <hip/hip_runtime.h>

typedef __attribute__((ext_vector_type(16))) _Float16 v16h;
typedef __attribute__((ext_vector_type(8)))  _Float16 v8h;
typedef __attribute__((ext_vector_type(8)))  float    v8f;
typedef __attribute__((ext_vector_type(4)))  float    v4f;

#define LTOK   64
#define EDIM   96
#define NHD    6
#define TPB    192   // 6 wave32
#define NWAVE  6

// A-operand pattern: lane gets contiguous K [hi*16 .. hi*16+15].
// Caller passes p = chunk_base + hi*16; elements 0..15 at p[0..15].
__device__ __forceinline__ v16h load_a_tile(const _Float16* p) {
  v8h lo = *(const v8h*)(p);
  v8h hh = *(const v8h*)(p + 8);
  v16h r;
#pragma unroll
  for (int e = 0; e < 8; ++e) { r[e] = lo[e]; r[e + 8] = hh[e]; }
  return r;
}

__global__ __launch_bounds__(TPB) void swin_attn_fused(
    const float* __restrict__ Q, const float* __restrict__ K, const float* __restrict__ V,
    const float* __restrict__ MASK, const float* __restrict__ BIAS,
    const float* __restrict__ GAMMA, const float* __restrict__ BETA,
    const float* __restrict__ PW, const float* __restrict__ PB,
    const int* __restrict__ ISM,
    float* __restrict__ OQ, float* __restrict__ OK, float* __restrict__ OV)
{
  // 112 KB LDS, manual layout; epilogue aliases dead attention buffers.
  __shared__ __align__(32) unsigned char smem[114688];
  _Float16* qhp  = (_Float16*)(smem);            // [64][192] q*scale, per-head 32-slot zero-padded
  _Float16* khp  = (_Float16*)(smem + 24576);    // [64][192] k, same padding
  _Float16* vT   = (_Float16*)(smem + 49152);    // [96][64]  v transposed (channel-major)
  _Float16* ps   = (_Float16*)(smem + 61440);    // [6 waves][16][64] unnormalized exp(P)
  float*    xs   = (float*)   (smem + 73728);    // [64][96] f32 shortcut + attn out
  float*    mterm= (float*)   (smem + 98304);    // [64][64] f32 mask term (diag + NEG folded)
  _Float16* lnh  = (_Float16*)(smem);            // phase-B alias: LN(x) f16 [64][96]
  _Float16* wt2  = (_Float16*)(smem + 12288);    // phase-B alias: W row-major f16 [96][96]
  float*    red  = (float*)   (smem + 61440);    // phase-B alias: LN reduce scratch (384 f32)

  const int tid  = threadIdx.x;
  const int wave = tid >> 5;
  const int lane = tid & 31;
  const int ml   = lane & 15;
  const int hi   = lane >> 4;
  const int w    = blockIdx.x;
  const size_t base = (size_t)w * (LTOK * EDIM);

  const int isM = ISM[0];
  const float* mk = MASK + (size_t)(w & 127) * (64 * 64);

  // ---- load phase ----
  const float4* q4 = (const float4*)(Q + base);
  const float4* k4 = (const float4*)(K + base);
  const float4* v4 = (const float4*)(V + base);
  float4* ok4 = (float4*)(OK + base);
  float4* ov4 = (float4*)(OV + base);
  float4* xs4 = (float4*)xs;

  for (int i = tid; i < (LTOK * EDIM) / 4; i += TPB) {
    float4 a = q4[i];
    float4 b = k4[i];
    float4 c = v4[i];
    xs4[i] = a;                  // f32 shortcut accumulator
    ok4[i] = b;                  // exact key passthrough
    ov4[i] = c;                  // exact value passthrough
    const int t  = i / 24;       // token (24 float4 per 96-ch row)
    const int c0 = (i - t * 24) * 4;         // channel base (multiple of 4)
    const int g  = c0 >> 4;                  // head group
    const int so = t * 192 + g * 32 + (c0 & 15);
    qhp[so + 0] = (_Float16)(a.x * 0.25f);   // SCALE = 16^-0.5
    qhp[so + 1] = (_Float16)(a.y * 0.25f);
    qhp[so + 2] = (_Float16)(a.z * 0.25f);
    qhp[so + 3] = (_Float16)(a.w * 0.25f);
    khp[so + 0] = (_Float16)b.x;
    khp[so + 1] = (_Float16)b.y;
    khp[so + 2] = (_Float16)b.z;
    khp[so + 3] = (_Float16)b.w;
    if ((c0 & 15) == 0) {        // zero this 16-group's pad half, exactly once
      v8h z = {};
      *(v8h*)(qhp + t * 192 + g * 32 + 16) = z;
      *(v8h*)(qhp + t * 192 + g * 32 + 24) = z;
      *(v8h*)(khp + t * 192 + g * 32 + 16) = z;
      *(v8h*)(khp + t * 192 + g * 32 + 24) = z;
    }
    vT[(c0 + 0) * 64 + t] = (_Float16)c.x;   // transposed V
    vT[(c0 + 1) * 64 + t] = (_Float16)c.y;
    vT[(c0 + 2) * 64 + t] = (_Float16)c.z;
    vT[(c0 + 3) * 64 + t] = (_Float16)c.w;
  }

  // mask term precompute: mterm[m][n] = where(mask'!=0, -1e9, mask'), diag folded
  for (int i = tid; i < (64 * 64) / 4; i += TPB) {
    float4 mv = ((const float4*)mk)[i];
    const int m  = i >> 4;          // 16 float4 per 64-wide row
    const int n0 = (i & 15) * 4;
    if (isM) {
      if (m == n0 + 0) mv.x = 1.0f;
      if (m == n0 + 1) mv.y = 1.0f;
      if (m == n0 + 2) mv.z = 1.0f;
      if (m == n0 + 3) mv.w = 1.0f;
    }
    float4 o;
    o.x = (mv.x != 0.0f) ? -1.0e9f : mv.x;
    o.y = (mv.y != 0.0f) ? -1.0e9f : mv.y;
    o.z = (mv.z != 0.0f) ? -1.0e9f : mv.z;
    o.w = (mv.w != 0.0f) ? -1.0e9f : mv.w;
    ((float4*)mterm)[i] = o;
  }
  __syncthreads();

  // all-ones A operand for the row-sum WMMA
  v16h onesv;
#pragma unroll
  for (int e = 0; e < 16; ++e) onesv[e] = (_Float16)1.0f;

  // ---- attention: 24 (head, m-tile) units, wave-uniform ----
  for (int u = wave; u < NHD * 4; u += NWAVE) {
    const int h  = u >> 2;
    const int mt = u & 3;

    // A = scale*Q tile [16 x 32(padded)] : two b128 LDS loads, no selects
    v16h aq = load_a_tile(qhp + (mt * 16 + ml) * 192 + h * 32 + hi * 16);

    // Affine bias base:  m = mt*16+8*hi+r ; n = nt*16+ml
    const float* bp = BIAS +
        ((((2 * mt + hi) - (ml >> 3) + 7) * 15 + (7 - (ml & 7))) * NHD + h);
    const float* mq = mterm + (mt * 16 + 8 * hi) * 64 + ml;

    v8f acc[4];
#pragma unroll
    for (int nt = 0; nt < 4; ++nt) {
      // C operand = bias + mask term, added for free inside the WMMA
      v8f cin;
#pragma unroll
      for (int r = 0; r < 8; ++r)
        cin[r] = bp[r * 6 - nt * 180] + mq[r * 64 + nt * 16];
      // B = K^T tile [32 x 16] in [n][K] layout = padded k rows: contiguous
      v16h bk = *(const v16h*)(khp + (nt * 16 + ml) * 192 + h * 32 + hi * 16);
      acc[nt] = __builtin_amdgcn_wmma_f32_16x16x32_f16(
          false, aq, false, bk, (short)0, cin, false, false);
    }

    // ---- tile-wide max (softmax is shift-invariant; logits are O(30)) ----
    float mr0 = fmaxf(fmaxf(acc[0][0], acc[1][0]), fmaxf(acc[2][0], acc[3][0]));
#pragma unroll
    for (int r = 1; r < 8; ++r) {
      float m3 = fmaxf(fmaxf(acc[0][r], acc[1][r]), fmaxf(acc[2][r], acc[3][r]));
      mr0 = fmaxf(mr0, m3);
    }
    mr0 = fmaxf(mr0, __shfl_xor(mr0, 1));
    mr0 = fmaxf(mr0, __shfl_xor(mr0, 2));
    mr0 = fmaxf(mr0, __shfl_xor(mr0, 4));
    mr0 = fmaxf(mr0, __shfl_xor(mr0, 8));
    mr0 = fmaxf(mr0, __shfl_xor(mr0, 16));

    // ---- exp (unnormalized, all <= 1 -> ideal f16 range) -> per-wave LDS P ----
    _Float16* pwv = ps + wave * (16 * 64);
#pragma unroll
    for (int r = 0; r < 8; ++r)
#pragma unroll
      for (int nt = 0; nt < 4; ++nt)
        pwv[(r + 8 * hi) * 64 + nt * 16 + ml] = (_Float16)__expf(acc[nt][r] - mr0);

    // ---- O^T = V^T @ P^T and rowsum = ONES @ P^T (shared B operand) ----
    v8f o = {}, srow = {};
#pragma unroll
    for (int kc = 0; kc < 2; ++kc) {
      v16h av  = load_a_tile(vT + (h * 16 + ml) * 64 + kc * 32 + hi * 16);
      v16h bpm = *(const v16h*)(pwv + ml * 64 + kc * 32 + hi * 16);
      o    = __builtin_amdgcn_wmma_f32_16x16x32_f16(
          false, av, false, bpm, (short)0, o, false, false);
      srow = __builtin_amdgcn_wmma_f32_16x16x32_f16(
          false, onesv, false, bpm, (short)0, srow, false, false);
    }
    // C layout of O^T: c[r] = O[m=ml][ch=h*16+8*hi+r]; srow[r] = rowsum[ml] (any r)
    const float inv = 1.0f / srow[0];
    float* xrow = xs + (mt * 16 + ml) * EDIM + h * 16 + 8 * hi;
    v4f x0 = *(v4f*)(xrow);
    v4f x1 = *(v4f*)(xrow + 4);
#pragma unroll
    for (int r = 0; r < 4; ++r) { x0[r] += o[r] * inv; x1[r] += o[r + 4] * inv; }
    *(v4f*)(xrow)     = x0;
    *(v4f*)(xrow + 4) = x1;
  }
  __syncthreads();

  // ---- phase B: W (row-major, coalesced) -> f16 LDS; parallel layernorm ----
  const float4* pw4 = (const float4*)PW;
  for (int i = tid; i < (EDIM * EDIM) / 4; i += TPB) {
    float4 wv = pw4[i];
    const int j = i * 4;
    wt2[j + 0] = (_Float16)wv.x;
    wt2[j + 1] = (_Float16)wv.y;
    wt2[j + 2] = (_Float16)wv.z;
    wt2[j + 3] = (_Float16)wv.w;
  }
  {
    const int t3 = tid & 63;          // token
    const int th = tid >> 6;          // third: channels [th*32, th*32+32)
    float s1 = 0.f, s2 = 0.f;
#pragma unroll
    for (int c = 0; c < 32; ++c) {
      float x = xs[t3 * EDIM + th * 32 + c];
      s1 += x;
      s2 += x * x;
    }
    red[tid]       = s1;
    red[192 + tid] = s2;
    __syncthreads();
    float sum  = red[t3] + red[64 + t3] + red[128 + t3];
    float sumq = red[192 + t3] + red[256 + t3] + red[320 + t3];
    float mu   = sum * (1.0f / EDIM);
    float var  = sumq * (1.0f / EDIM) - mu * mu;
    float inv  = rsqrtf(var + 1e-5f);
#pragma unroll
    for (int c = 0; c < 32; ++c) {
      const int cc = th * 32 + c;
      lnh[t3 * EDIM + cc] =
          (_Float16)((xs[t3 * EDIM + cc] - mu) * inv * GAMMA[cc] + BETA[cc]);
    }
  }
  __syncthreads();

  // ---- projection: y = LN(x) @ W^T + b ; 24 tiles, K=96 in 3 chunks ----
  for (int u = wave; u < 24; u += NWAVE) {
    const int ot = u >> 2;
    const int mt = u & 3;
    v8f acc2 = {};
#pragma unroll
    for (int ks = 0; ks < 3; ++ks) {
      v16h a = load_a_tile(lnh + (mt * 16 + ml) * EDIM + ks * 32 + hi * 16);
      // B[k=e][n=o] = W[o][e] -> [n][K] layout = row-major W: contiguous
      v16h b = *(const v16h*)(wt2 + (ot * 16 + ml) * EDIM + ks * 32 + hi * 16);
      acc2 = __builtin_amdgcn_wmma_f32_16x16x32_f16(
          false, a, false, b, (short)0, acc2, false, false);
    }
    float pb = PB[ot * 16 + ml];
#pragma unroll
    for (int r = 0; r < 8; ++r)
      OQ[base + (size_t)(mt * 16 + r + 8 * hi) * EDIM + ot * 16 + ml] = acc2[r] + pb;
  }
}

extern "C" void kernel_launch(void* const* d_in, const int* in_sizes, int n_in,
                              void* d_out, int out_size, void* d_ws, size_t ws_size,
                              hipStream_t stream) {
  const float* Q     = (const float*)d_in[0];
  const float* K     = (const float*)d_in[1];
  const float* V     = (const float*)d_in[2];
  const float* MASK  = (const float*)d_in[3];
  const float* BIAS  = (const float*)d_in[4];
  const float* GAMMA = (const float*)d_in[5];
  const float* BETA  = (const float*)d_in[6];
  const float* PW    = (const float*)d_in[7];
  const float* PB    = (const float*)d_in[8];
  const int*   ISM   = (const int*)d_in[9];

  const size_t NT = (size_t)8192 * LTOK * EDIM;
  float* OQ = (float*)d_out;
  float* OK = OQ + NT;
  float* OV = OK + NT;

  swin_attn_fused<<<8192, TPB, 0, stream>>>(Q, K, V, MASK, BIAS, GAMMA, BETA,
                                            PW, PB, ISM, OQ, OK, OV);
}
